// QuantizedLinear_3427383902219
// MI455X (gfx1250) — compile-verified, matrix operand order checked by reading, not run on hardware
//
#include <hip/hip_runtime.h>

// ---------------------------------------------------------------------------
// QuantizedLinear on MI455X (gfx1250, wave32, WMMA):
//   y[m,o] = scale[o] * dot(x[m,:], q[o,:]) - scale[o]*zp[o]*rowsum[m] + bias[o]
// bf16 WMMA (int8 weights are EXACT in bf16), f32 accumulate, per-channel
// dequant folded into the f32 epilogue. Ping-pong LDS double buffering with
// register staging: 1 barrier per K-step, global latency hidden under WMMA.
// ---------------------------------------------------------------------------

#define TM 128      // block tile M
#define TN 128      // block tile N
#define KSTEP 64    // K per LDS stage (2 WMMA-K chunks of 32)
#define LDA 72      // padded LDS row stride (bf16 elems): 144B, 16B-aligned
#define LDB 72

typedef __attribute__((ext_vector_type(16))) __bf16 v16bf;
typedef __attribute__((ext_vector_type(8)))  __bf16 v8bf;
typedef __attribute__((ext_vector_type(8)))  float  v8f;

union Frag {
    v16bf v;
    v8bf  h[2];
};

// Per-thread register staging for one (TM x KSTEP) A tile + (TN x KSTEP) B tile:
// 256 threads -> 32 x-floats (8 float4) and 32 w-bytes (2 int4) per thread.
struct StageRegs {
    float4 xf[8];
    int4   wb[2];
};

// ---------------------------------------------------------------------------
// Pre-kernel: rowsum[m] = sum_k x[m,k]  (for the zero-point term)
// ---------------------------------------------------------------------------
__global__ __launch_bounds__(256) void rowsum_kernel(const float* __restrict__ x,
                                                     float* __restrict__ rowsum,
                                                     int K) {
    __shared__ float red[256];
    const int m = blockIdx.x;
    const float* row = x + (size_t)m * K;
    float s = 0.f;
    for (int k = threadIdx.x; k < K; k += 256) s += row[k];
    red[threadIdx.x] = s;
    __syncthreads();
    for (int off = 128; off > 0; off >>= 1) {
        if (threadIdx.x < off) red[threadIdx.x] += red[threadIdx.x + off];
        __syncthreads();
    }
    if (threadIdx.x == 0) rowsum[m] = red[0];
}

// ---------------------------------------------------------------------------
// Main GEMM: 256 threads = 8 waves; block tile 128x128; wave tile 32x64.
// ---------------------------------------------------------------------------
__global__ __launch_bounds__(256) void qlinear_wmma_kernel(
    const float* __restrict__ x,          // [M,K] fp32
    const signed char* __restrict__ wq,   // [N,K] int8
    const float* __restrict__ scale,      // [N]
    const float* __restrict__ zp,         // [N]
    const float* __restrict__ bias,       // [N]
    const float* __restrict__ rowsum,     // [M]
    float* __restrict__ out,              // [M,N] fp32
    int M, int N, int K)
{
    // ping-pong double buffers
    __shared__ __align__(16) __bf16 As[2][TM * LDA];
    __shared__ __align__(16) __bf16 Bs[2][TN * LDB];

    const int t     = threadIdx.x;
    const int lane  = t & 31;
    const int wave  = t >> 5;        // 0..7
    const int wm    = wave >> 1;     // 0..3 : wave row (M)
    const int wn    = wave & 1;      // 0..1 : wave col (N)
    const int lrow  = lane & 15;     // row within 16x16 fragment
    const int lhalf = lane >> 4;     // 0: K-halves {0-7,16-23}, 1: {8-15,24-31}

    const int m_base = blockIdx.y * TM;
    const int n_base = blockIdx.x * TN;

    // staging-load mapping: 2 threads per row, 32 elems each
    const int lr = t >> 1;            // 0..127
    const int lc = (t & 1) * 32;      // 0 or 32

    const float*       xp = x  + (size_t)(m_base + lr) * K + lc;  // advance by KSTEP
    const signed char* wp = wq + (size_t)(n_base + lr) * K + lc;

    v8f acc[2][4];
#pragma unroll
    for (int mi = 0; mi < 2; ++mi)
#pragma unroll
        for (int ni = 0; ni < 4; ++ni)
            acc[mi][ni] = v8f{};

    StageRegs r;

    // ---- prologue: stage tile 0 --------------------------------------------
#pragma unroll
    for (int i = 0; i < 8; ++i) r.xf[i] = *(const float4*)(xp + i * 4);
#pragma unroll
    for (int i = 0; i < 2; ++i) r.wb[i] = *(const int4*)(wp + i * 16);

    {
        __bf16* ap = &As[0][lr * LDA + lc];
#pragma unroll
        for (int i = 0; i < 4; ++i) {
            v8bf h;
            h[0] = (__bf16)r.xf[2 * i].x;     h[1] = (__bf16)r.xf[2 * i].y;
            h[2] = (__bf16)r.xf[2 * i].z;     h[3] = (__bf16)r.xf[2 * i].w;
            h[4] = (__bf16)r.xf[2 * i + 1].x; h[5] = (__bf16)r.xf[2 * i + 1].y;
            h[6] = (__bf16)r.xf[2 * i + 1].z; h[7] = (__bf16)r.xf[2 * i + 1].w;
            *(v8bf*)(ap + i * 8) = h;
        }
        __bf16* bp = &Bs[0][lr * LDB + lc];
#pragma unroll
        for (int i = 0; i < 2; ++i) {
            union { int4 q; signed char b[16]; } u;
            u.q = r.wb[i];
            v8bf h0, h1;
#pragma unroll
            for (int j = 0; j < 8; ++j) h0[j] = (__bf16)(float)u.b[j];
#pragma unroll
            for (int j = 0; j < 8; ++j) h1[j] = (__bf16)(float)u.b[j + 8];
            *(v8bf*)(bp + i * 16)     = h0;
            *(v8bf*)(bp + i * 16 + 8) = h1;
        }
    }

    // ---- main pipelined K loop ---------------------------------------------
    int buf = 0;
    for (int k0 = 0; k0 < K; k0 += KSTEP, buf ^= 1) {
        __syncthreads();   // lds[buf] ready; lds[buf^1] free to overwrite

        const bool more = (k0 + KSTEP) < K;
        if (more) {
            // issue next tile's global loads now; latency hidden under WMMAs
            const float*       xn = xp + (k0 + KSTEP);
            const signed char* wn_ = wp + (k0 + KSTEP);
#pragma unroll
            for (int i = 0; i < 8; ++i) r.xf[i] = *(const float4*)(xn + i * 4);
#pragma unroll
            for (int i = 0; i < 2; ++i) r.wb[i] = *(const int4*)(wn_ + i * 16);
        }

        // ---- compute: 2 K-chunks of 32, 2x4 WMMA tiles per wave ------------
        const __bf16* Ab = As[buf];
        const __bf16* Bb = Bs[buf];
#pragma unroll
        for (int c = 0; c < 2; ++c) {
            Frag a[2], b[4];
#pragma unroll
            for (int mi = 0; mi < 2; ++mi) {
                const __bf16* base =
                    &Ab[(wm * 32 + mi * 16 + lrow) * LDA + c * 32 + lhalf * 8];
                a[mi].h[0] = *(const v8bf*)base;         // K 0-7  (or 8-15)
                a[mi].h[1] = *(const v8bf*)(base + 16);  // K 16-23 (or 24-31)
            }
#pragma unroll
            for (int ni = 0; ni < 4; ++ni) {
                const __bf16* base =
                    &Bb[(wn * 64 + ni * 16 + lrow) * LDB + c * 32 + lhalf * 8];
                b[ni].h[0] = *(const v8bf*)base;
                b[ni].h[1] = *(const v8bf*)(base + 16);
            }
#pragma unroll
            for (int mi = 0; mi < 2; ++mi)
#pragma unroll
                for (int ni = 0; ni < 4; ++ni)
                    acc[mi][ni] = __builtin_amdgcn_wmma_f32_16x16x32_bf16(
                        false, a[mi].v, false, b[ni].v,
                        (short)0, acc[mi][ni], false, false);
        }

        // ---- convert + store next tile into the other buffer ---------------
        if (more) {
            __bf16* ap = &As[buf ^ 1][lr * LDA + lc];
#pragma unroll
            for (int i = 0; i < 4; ++i) {
                v8bf h;
                h[0] = (__bf16)r.xf[2 * i].x;     h[1] = (__bf16)r.xf[2 * i].y;
                h[2] = (__bf16)r.xf[2 * i].z;     h[3] = (__bf16)r.xf[2 * i].w;
                h[4] = (__bf16)r.xf[2 * i + 1].x; h[5] = (__bf16)r.xf[2 * i + 1].y;
                h[6] = (__bf16)r.xf[2 * i + 1].z; h[7] = (__bf16)r.xf[2 * i + 1].w;
                *(v8bf*)(ap + i * 8) = h;
            }
            __bf16* bp = &Bs[buf ^ 1][lr * LDB + lc];
#pragma unroll
            for (int i = 0; i < 2; ++i) {
                union { int4 q; signed char b[16]; } u;
                u.q = r.wb[i];
                v8bf h0, h1;
#pragma unroll
                for (int j = 0; j < 8; ++j) h0[j] = (__bf16)(float)u.b[j];
#pragma unroll
                for (int j = 0; j < 8; ++j) h1[j] = (__bf16)(float)u.b[j + 8];
                *(v8bf*)(bp + i * 16)     = h0;
                *(v8bf*)(bp + i * 16 + 8) = h1;
            }
        }
    }

    // ---- epilogue: per-channel dequant + zero-point + bias, f32 store -------
    // C/D layout: VGPR r -> (lanes 0-15: M=r, N=lane), (lanes 16-31: M=r+8, N=lane-16)
#pragma unroll
    for (int ni = 0; ni < 4; ++ni) {
        const int n = n_base + wn * 64 + ni * 16 + lrow;
        const float s  = scale[n];
        const float bb = bias[n];
        const float sz = s * zp[n];
#pragma unroll
        for (int mi = 0; mi < 2; ++mi) {
            const int mrow = m_base + wm * 32 + mi * 16 + lhalf * 8;
#pragma unroll
            for (int rr = 0; rr < 8; ++rr) {
                const int m = mrow + rr;
                const float v = acc[mi][ni][rr] * s + (bb - sz * rowsum[m]);
                out[(size_t)m * N + n] = v;
            }
        }
    }
}

// ---------------------------------------------------------------------------
extern "C" void kernel_launch(void* const* d_in, const int* in_sizes, int n_in,
                              void* d_out, int out_size, void* d_ws, size_t ws_size,
                              hipStream_t stream) {
    const float*       x     = (const float*)d_in[0];
    const signed char* wq    = (const signed char*)d_in[1];
    const float*       scale = (const float*)d_in[2];
    const float*       zp    = (const float*)d_in[3];
    const float*       bias  = (const float*)d_in[4];
    float*             out   = (float*)d_out;
    float*             rsum  = (float*)d_ws;   // M floats of scratch

    const int M = 4 * 2048;   // B*S
    const int N = 16384;      // OUT
    const int K = 4096;       // IN

    rowsum_kernel<<<M, 256, 0, stream>>>(x, rsum, K);

    dim3 grid(N / TN, M / TM);   // 128 x 64 blocks
    qlinear_wmma_kernel<<<grid, 256, 0, stream>>>(x, wq, scale, zp, bias, rsum,
                                                  out, M, N, K);
}
